// Net_90675349553198
// MI455X (gfx1250) — matrix operand order repaired and checked by reading, not compiled, not run on hardware
//
#include <hip/hip_runtime.h>
#include <cstdint>
#include <cstddef>

typedef _Float16 v16h __attribute__((ext_vector_type(16)));
typedef _Float16 v8h  __attribute__((ext_vector_type(8)));
typedef float    v8f  __attribute__((ext_vector_type(8)));

#define NWAVES 8
#define BN_SC 0.9999950000374997f   /* 1/sqrt(1+1e-5) */

// ---------------------------------------------------------------------------
// Weight packing: f32 [Cin][Cout] -> f16 fragment-major layout for WMMA B:
//   dst[(((nt*ksteps)+ks)*32 + lane)*16 + i] = W[ks*32 + (i>>3)*16 + (lane>>4)*8 + (i&7)][nt*16 + (lane&15)]
// (zero-padded for k >= Cin). KdimPad % 32 == 0, Cout % 16 == 0.
// Each lane's 16 halves are contiguous & 32B aligned -> 2x global_load_b128.
// ---------------------------------------------------------------------------
__global__ void pack_w_kernel(const float* __restrict__ W, _Float16* __restrict__ dst,
                              int Cin, int Cout, int KdimPad) {
    int idx = blockIdx.x * blockDim.x + threadIdx.x;
    int total = KdimPad * Cout;
    if (idx >= total) return;
    int ksteps = KdimPad >> 5;
    int i    = idx & 15;
    int lane = (idx >> 4) & 31;
    int rest = idx >> 9;
    int ks   = rest % ksteps;
    int nt   = rest / ksteps;
    int col  = (nt << 4) + (lane & 15);
    int kb   = (ks << 5) + ((i >> 3) << 4) + ((lane >> 4) << 3) + (i & 7);
    dst[idx] = (kb < Cin) ? (_Float16)W[(size_t)kb * Cout + col] : (_Float16)0.0f;
}

// ---------------------------------------------------------------------------
// Farthest point sampling. One block per batch, N threads (N power of two).
// Start index 0; argmax ties -> lowest index (matches jnp.argmax).
// ---------------------------------------------------------------------------
template <int N>
__global__ void fps_kernel(const float* __restrict__ pos, int S, float* __restrict__ newpos) {
    int b = blockIdx.x, t = threadIdx.x;
    __shared__ float rd[N];
    __shared__ int   ri[N];
    __shared__ float f[3];
    const float px = pos[((size_t)b * N + t) * 3 + 0];
    const float py = pos[((size_t)b * N + t) * 3 + 1];
    const float pz = pos[((size_t)b * N + t) * 3 + 2];
    float dist = __builtin_inff();
    int far = 0;
    for (int i = 0; i < S; ++i) {
        if (t == far) {
            f[0] = px; f[1] = py; f[2] = pz;
            newpos[((size_t)b * S + i) * 3 + 0] = px;
            newpos[((size_t)b * S + i) * 3 + 1] = py;
            newpos[((size_t)b * S + i) * 3 + 2] = pz;
        }
        __syncthreads();
        float dx = px - f[0], dy = py - f[1], dz = pz - f[2];
        float d = dx * dx + dy * dy + dz * dz;
        dist = fminf(dist, d);
        rd[t] = dist; ri[t] = t;
        __syncthreads();
        for (int off = N >> 1; off > 0; off >>= 1) {
            if (t < off) {
                float dv = rd[t + off]; int iv = ri[t + off];
                if (dv > rd[t] || (dv == rd[t] && iv < ri[t])) { rd[t] = dv; ri[t] = iv; }
            }
            __syncthreads();
        }
        far = ri[0];
        __syncthreads();
    }
}

// ---------------------------------------------------------------------------
// Radius neighbor query (up to K=64 nearest within r). Block per batch,
// thread per sample. Matches top_k semantics: if >64 within radius, pick the
// 64 with lexicographically smallest (d2, idx).
// ---------------------------------------------------------------------------
template <int N>
__global__ void nq_kernel(const float* __restrict__ pos, const float* __restrict__ q,
                          int S, float r2, int* __restrict__ nidx, int* __restrict__ cnt) {
    int b = blockIdx.x, s = threadIdx.x;
    __shared__ float sp[N * 3];
    for (int i = s; i < N * 3; i += blockDim.x) sp[i] = pos[(size_t)b * N * 3 + i];
    __syncthreads();
    const float qx = q[((size_t)b * S + s) * 3 + 0];
    const float qy = q[((size_t)b * S + s) * 3 + 1];
    const float qz = q[((size_t)b * S + s) * 3 + 2];
    int c = 0;
    for (int j = 0; j < N; ++j) {
        float dx = sp[j * 3 + 0] - qx, dy = sp[j * 3 + 1] - qy, dz = sp[j * 3 + 2] - qz;
        float d2 = dx * dx + dy * dy + dz * dz;
        if (d2 <= r2) ++c;
    }
    int* out = nidx + ((size_t)b * S + s) * 64;
    if (c <= 64) {
        int k = 0;
        for (int j = 0; j < N; ++j) {
            float dx = sp[j * 3 + 0] - qx, dy = sp[j * 3 + 1] - qy, dz = sp[j * 3 + 2] - qz;
            float d2 = dx * dx + dy * dy + dz * dz;
            if (d2 <= r2) out[k++] = j;
        }
        for (; k < 64; ++k) out[k] = 0;
        cnt[(size_t)b * S + s] = c;
    } else {
        unsigned mask[(N + 31) / 32];
        for (int i = 0; i < (N + 31) / 32; ++i) mask[i] = 0u;
        for (int k = 0; k < 64; ++k) {
            float bd = __builtin_inff(); int bi = N;
            for (int j = 0; j < N; ++j) {
                if ((mask[j >> 5] >> (j & 31)) & 1u) continue;
                float dx = sp[j * 3 + 0] - qx, dy = sp[j * 3 + 1] - qy, dz = sp[j * 3 + 2] - qz;
                float d2 = dx * dx + dy * dy + dz * dz;
                if (d2 > r2) continue;
                if (d2 < bd || (d2 == bd && j < bi)) { bd = d2; bi = j; }
            }
            mask[bi >> 5] |= (1u << (bi & 31));
            out[k] = bi;
        }
        cnt[(size_t)b * S + s] = 64;
    }
}

// ---------------------------------------------------------------------------
// One MLP layer via WMMA f32<-f16: dst = relu(src@W + b) * (g/sqrt(1+eps)) + be
// A frag (16x32 f16): lane l: row = m0+(l&15); halves 0..7 = K kg..kg+7,
//   halves 8..15 = K 16+kg..16+kg+7, kg = 8*(l>>4)      (ISA 05_wmma layout)
// B frag: packed fragment-major by pack_w_kernel -> one contiguous v16h/lane.
// D (f32): lane l: col = n0+(l&15); vgpr j -> row = m0 + j + 8*(l>>4)
// ---------------------------------------------------------------------------
template <int ROWS>
__device__ inline void gemm_layer(const _Float16* src, int strideS, int Kdim,
                                  const _Float16* __restrict__ Wp, int Cout,
                                  const float* __restrict__ bias,
                                  const float* __restrict__ gam,
                                  const float* __restrict__ bet,
                                  _Float16* dst, int strideD,
                                  bool last, int cnt, unsigned* smax) {
    const int tid  = threadIdx.x;
    const int w    = tid >> 5;
    const int lane = tid & 31;
    const int lrow = lane & 15;
    const int lhi  = lane >> 4;
    const int tilesM = ROWS / 16;
    const int T = tilesM * (Cout / 16);
    const int ksteps = Kdim >> 5;
    for (int t = w; t < T; t += NWAVES) {
        const int m0 = (t % tilesM) * 16;
        const int n0 = (t / tilesM) * 16;
        v8f acc = {0.f, 0.f, 0.f, 0.f, 0.f, 0.f, 0.f, 0.f};
        const int arow = m0 + lrow;
        const int col  = n0 + lrow;
        const _Float16* wbase = Wp + ((((size_t)(n0 >> 4) * ksteps) << 9) | ((size_t)lane << 4));
        for (int kk = 0; kk < Kdim; kk += 32) {
            // A: two aligned 16B LDS chunks
            const v8h* ap = (const v8h*)(src + (size_t)arow * strideS + kk + (lhi << 3));
            v8h alo = ap[0];
            v8h ahi = ap[2];
            v16h a = __builtin_shufflevector(alo, ahi, 0, 1, 2, 3, 4, 5, 6, 7,
                                             8, 9, 10, 11, 12, 13, 14, 15);
            // B: one contiguous 32B aligned fragment per lane
            v16h bf = *(const v16h*)(wbase + ((size_t)(kk >> 5) << 9));
            acc = __builtin_amdgcn_wmma_f32_16x16x32_f16(false, a, false, bf,
                                                         (short)0, acc, false, false);
        }
        const float bb = bias[col];
        const float sc = gam[col] * BN_SC;
        const float bt = bet[col];
#pragma unroll
        for (int j = 0; j < 8; ++j) {
            const int row = m0 + j + (lhi << 3);
            float v = fmaxf(acc[j] + bb, 0.f) * sc + bt;
            if (last) {
                if (row < cnt) {
                    unsigned u = __float_as_uint(v);
                    u = (u & 0x80000000u) ? ~u : (u | 0x80000000u);
                    atomicMax(&smax[col], u);
                }
            } else {
                dst[(size_t)row * strideD + col] = (_Float16)v;
            }
        }
    }
}

// ---------------------------------------------------------------------------
// Fused PointConv: gather neighbors -> 3-layer MLP (WMMA) -> masked max.
// GATHER: block = (b,s); rows = 64 neighbors. !GATHER: block = b; rows = S3.
// ---------------------------------------------------------------------------
template <int ROWS, int CFEAT, int CINPAD, int C1, int C2, int C3, bool GATHER>
__global__ __launch_bounds__(256)
void pointconv_kernel(const float* __restrict__ xin, const float* __restrict__ posAll,
                      const float* __restrict__ qpos, const int* __restrict__ nidx,
                      const int* __restrict__ cnts,
                      const _Float16* __restrict__ W1, const _Float16* __restrict__ W2,
                      const _Float16* __restrict__ W3,
                      const float* __restrict__ b1, const float* __restrict__ g1, const float* __restrict__ be1,
                      const float* __restrict__ b2, const float* __restrict__ g2, const float* __restrict__ be2,
                      const float* __restrict__ b3, const float* __restrict__ g3, const float* __restrict__ be3,
                      int S, int N, float* __restrict__ out) {
    constexpr int MAXA = (CINPAD > C2) ? CINPAD : C2;
    __shared__ _Float16 act0[ROWS * MAXA];
    __shared__ _Float16 act1[ROWS * C1];
    __shared__ unsigned smax[C3];

    const int tid = threadIdx.x;
    const int bs  = blockIdx.x;
    const int b   = GATHER ? (bs / S) : bs;
    int cnt = ROWS;
    float qx = 0.f, qy = 0.f, qz = 0.f;
    const int* nb = nullptr;
    if constexpr (GATHER) {
        cnt = cnts[bs];
        qx = qpos[(size_t)bs * 3 + 0];
        qy = qpos[(size_t)bs * 3 + 1];
        qz = qpos[(size_t)bs * 3 + 2];
        nb = nidx + (size_t)bs * 64;
    }
    for (int c = tid; c < C3; c += 256) smax[c] = 0u;

    // Build A0 [ROWS, CINPAD] in f16
    for (int e = tid; e < ROWS * CINPAD; e += 256) {
        const int row = e / CINPAD, c = e - row * CINPAD;
        int src;
        if constexpr (GATHER) src = nb[row]; else src = row;
        float v = 0.f;
        if (c < CFEAT) {
            v = xin[((size_t)b * N + src) * CFEAT + c];
        } else if (c < CFEAT + 3) {
            float p = posAll[((size_t)b * N + src) * 3 + (c - CFEAT)];
            if constexpr (GATHER)
                v = p - ((c == CFEAT) ? qx : (c == CFEAT + 1) ? qy : qz);
            else
                v = p;
        }
        act0[(size_t)row * CINPAD + c] = (_Float16)v;
    }
    __syncthreads();

    gemm_layer<ROWS>(act0, CINPAD, CINPAD, W1, C1, b1, g1, be1, act1, C1, false, cnt, nullptr);
    __syncthreads();
    gemm_layer<ROWS>(act1, C1, C1, W2, C2, b2, g2, be2, act0, C2, false, cnt, nullptr);
    __syncthreads();
    gemm_layer<ROWS>(act0, C2, C2, W3, C3, b3, g3, be3, nullptr, 0, true, cnt, smax);
    __syncthreads();

    float* o = out + (size_t)bs * C3;
    for (int c = tid; c < C3; c += 256) {
        unsigned u = smax[c];
        u = (u & 0x80000000u) ? (u & 0x7FFFFFFFu) : ~u;
        o[c] = __uint_as_float(u);
    }
}

// ---------------------------------------------------------------------------
// Head MLP: [1024+8] -> 512 -> 256 -> 128 -> 2, ReLU between, log_softmax.
// One block per batch, fp32 scalar (tiny).
// ---------------------------------------------------------------------------
__global__ __launch_bounds__(256)
void head_kernel(const float* __restrict__ g, const float* __restrict__ glob,
                 const float* __restrict__ W1, const float* __restrict__ B1,
                 const float* __restrict__ W2, const float* __restrict__ B2,
                 const float* __restrict__ W3, const float* __restrict__ B3,
                 const float* __restrict__ W4, const float* __restrict__ B4,
                 float* __restrict__ out) {
    const int b = blockIdx.x, tid = threadIdx.x;
    __shared__ float s_in[1032];
    __shared__ float h1[512];
    __shared__ float h2[256];
    __shared__ float h3[128];
    __shared__ float lg[2];
    for (int i = tid; i < 1024; i += 256) s_in[i] = g[(size_t)b * 1024 + i];
    if (tid < 8) s_in[1024 + tid] = glob[(size_t)b * 8 + tid];
    __syncthreads();
    for (int c = tid; c < 512; c += 256) {
        float a = B1[c];
        for (int k = 0; k < 1032; ++k) a += s_in[k] * W1[(size_t)k * 512 + c];
        h1[c] = fmaxf(a, 0.f);
    }
    __syncthreads();
    if (tid < 256) {
        float a = B2[tid];
        for (int k = 0; k < 512; ++k) a += h1[k] * W2[(size_t)k * 256 + tid];
        h2[tid] = fmaxf(a, 0.f);
    }
    __syncthreads();
    if (tid < 128) {
        float a = B3[tid];
        for (int k = 0; k < 256; ++k) a += h2[k] * W3[(size_t)k * 128 + tid];
        h3[tid] = fmaxf(a, 0.f);
    }
    __syncthreads();
    if (tid < 2) {
        float a = B4[tid];
        for (int k = 0; k < 128; ++k) a += h3[k] * W4[(size_t)k * 2 + tid];
        lg[tid] = a;
    }
    __syncthreads();
    if (tid == 0) {
        float m = fmaxf(lg[0], lg[1]);
        float l = m + logf(expf(lg[0] - m) + expf(lg[1] - m));
        out[b * 2 + 0] = lg[0] - l;
        out[b * 2 + 1] = lg[1] - l;
    }
}

// ---------------------------------------------------------------------------
extern "C" void kernel_launch(void* const* d_in, const int* in_sizes, int n_in,
                              void* d_out, int out_size, void* d_ws, size_t ws_size,
                              hipStream_t stream) {
    (void)in_sizes; (void)n_in; (void)out_size; (void)ws_size;
    const float* x    = (const float*)d_in[0];   // [16,512,128]
    const float* pos  = (const float*)d_in[1];   // [16,512,3]
    const float* glob = (const float*)d_in[2];   // [16,8]
    // param bases: sa1=3, sa1a=15, sa2=27, sa3=39, head=51

    // ---- workspace carving ----
    char* wsB = (char*)d_ws;
    size_t off = 0;
    auto alloc = [&](size_t bytes) -> void* {
        void* p = wsB + off;
        off = (off + bytes + 255) & ~(size_t)255;
        return p;
    };
    // f16 packed weights (12 layers): {Cin, CinPad, Cout}
    const int lw[12][3] = {
        {131, 160, 128}, {128, 128, 128}, {128, 128, 160},   // sa1
        {163, 192, 160}, {160, 160, 160}, {160, 160, 192},   // sa1a
        {195, 224, 192}, {192, 192, 192}, {192, 192, 256},   // sa2
        {259, 288, 256}, {256, 256, 512}, {512, 512, 1024}}; // sa3
    const int lbase[12] = {3, 7, 11, 15, 19, 23, 27, 31, 35, 39, 43, 47};
    _Float16* wf[12];
    for (int i = 0; i < 12; ++i)
        wf[i] = (_Float16*)alloc((size_t)lw[i][1] * lw[i][2] * sizeof(_Float16));

    float* np1   = (float*)alloc((size_t)16 * 256 * 3 * 4);
    float* np2   = (float*)alloc((size_t)16 * 128 * 3 * 4);
    float* np3   = (float*)alloc((size_t)16 * 32 * 3 * 4);
    int*   nidx1 = (int*)alloc((size_t)16 * 256 * 64 * 4);
    int*   cnt1  = (int*)alloc((size_t)16 * 256 * 4);
    int*   nidx2 = (int*)alloc((size_t)16 * 128 * 64 * 4);
    int*   cnt2  = (int*)alloc((size_t)16 * 128 * 4);
    int*   nidx3 = (int*)alloc((size_t)16 * 32 * 64 * 4);
    int*   cnt3  = (int*)alloc((size_t)16 * 32 * 4);
    float* x1    = (float*)alloc((size_t)16 * 256 * 160 * 4);
    float* x2    = (float*)alloc((size_t)16 * 128 * 192 * 4);
    float* x3    = (float*)alloc((size_t)16 * 32 * 256 * 4);
    float* gbuf  = (float*)alloc((size_t)16 * 1024 * 4);

    // ---- pack weights into WMMA fragment-major f16 layout ----
    for (int i = 0; i < 12; ++i) {
        int total = lw[i][1] * lw[i][2];
        pack_w_kernel<<<(total + 255) / 256, 256, 0, stream>>>(
            (const float*)d_in[lbase[i]], wf[i], lw[i][0], lw[i][2], lw[i][1]);
    }

    auto P = [&](int base, int layer, int which) -> const float* {
        return (const float*)d_in[base + layer * 4 + which];
    };

    // ---- stage sa1: 512 -> 256 samples, r=0.2, MLP 131->128->128->160 ----
    fps_kernel<512><<<16, 512, 0, stream>>>(pos, 256, np1);
    nq_kernel<512><<<16, 256, 0, stream>>>(pos, np1, 256, 0.04f, nidx1, cnt1);
    pointconv_kernel<64, 128, 160, 128, 128, 160, true><<<16 * 256, 256, 0, stream>>>(
        x, pos, np1, nidx1, cnt1, wf[0], wf[1], wf[2],
        P(3, 0, 1), P(3, 0, 2), P(3, 0, 3),
        P(3, 1, 1), P(3, 1, 2), P(3, 1, 3),
        P(3, 2, 1), P(3, 2, 2), P(3, 2, 3),
        256, 512, x1);

    // ---- stage sa1a: 256 -> 128, r=0.2, MLP 163->160->160->192 ----
    fps_kernel<256><<<16, 256, 0, stream>>>(np1, 128, np2);
    nq_kernel<256><<<16, 128, 0, stream>>>(np1, np2, 128, 0.04f, nidx2, cnt2);
    pointconv_kernel<64, 160, 192, 160, 160, 192, true><<<16 * 128, 256, 0, stream>>>(
        x1, np1, np2, nidx2, cnt2, wf[3], wf[4], wf[5],
        P(15, 0, 1), P(15, 0, 2), P(15, 0, 3),
        P(15, 1, 1), P(15, 1, 2), P(15, 1, 3),
        P(15, 2, 1), P(15, 2, 2), P(15, 2, 3),
        128, 256, x2);

    // ---- stage sa2: 128 -> 32, r=0.4, MLP 195->192->192->256 ----
    fps_kernel<128><<<16, 128, 0, stream>>>(np2, 32, np3);
    nq_kernel<128><<<16, 32, 0, stream>>>(np2, np3, 32, 0.16f, nidx3, cnt3);
    pointconv_kernel<64, 192, 224, 192, 192, 256, true><<<16 * 32, 256, 0, stream>>>(
        x2, np2, np3, nidx3, cnt3, wf[6], wf[7], wf[8],
        P(27, 0, 1), P(27, 0, 2), P(27, 0, 3),
        P(27, 1, 1), P(27, 1, 2), P(27, 1, 3),
        P(27, 2, 1), P(27, 2, 2), P(27, 2, 3),
        32, 128, x3);

    // ---- global SA: MLP 259->256->512->1024, max over 32 points ----
    pointconv_kernel<32, 256, 288, 256, 512, 1024, false><<<16, 256, 0, stream>>>(
        x3, np3, nullptr, nullptr, nullptr, wf[9], wf[10], wf[11],
        P(39, 0, 1), P(39, 0, 2), P(39, 0, 3),
        P(39, 1, 1), P(39, 1, 2), P(39, 1, 3),
        P(39, 2, 1), P(39, 2, 2), P(39, 2, 3),
        1, 32, gbuf);

    // ---- head + log_softmax ----
    head_kernel<<<16, 256, 0, stream>>>(
        gbuf, glob,
        (const float*)d_in[51], (const float*)d_in[52],
        (const float*)d_in[53], (const float*)d_in[54],
        (const float*)d_in[55], (const float*)d_in[56],
        (const float*)d_in[57], (const float*)d_in[58],
        (float*)d_out);
}
